// FastAttention_36756330119890
// MI455X (gfx1250) — matrix operand order repaired and checked by reading, not compiled, not run on hardware
//
#include <hip/hip_runtime.h>

typedef __attribute__((ext_vector_type(16))) __bf16 v16bf;
typedef __attribute__((ext_vector_type(8)))  float  v8f;
typedef int v4i_vec __attribute__((vector_size(16)));   // matches async-LDS builtin param

#define Bdim   4
#define Ndim   8192
#define DIMd   1024
#define Hd     16
#define DHd    64
#define INNERd 1024
#define MASK_VAL (-1.0e9f)

#if defined(__has_builtin)
#if __has_builtin(__builtin_amdgcn_global_load_async_to_lds_b128)
#define HAVE_ASYNC 1
#endif
#endif
#ifndef HAVE_ASYNC
#define HAVE_ASYNC 0
#endif

struct __align__(16) Frag16 { unsigned short us[16]; };

__device__ __forceinline__ unsigned short f2bf(float f) {
    unsigned int u = __float_as_uint(f);
    u += 0x7FFFu + ((u >> 16) & 1u);
    return (unsigned short)(u >> 16);
}

__device__ __forceinline__ v8f v8f_zero() {
    v8f z;
#pragma unroll
    for (int i = 0; i < 8; ++i) z[i] = 0.f;
    return z;
}

// 16-byte global -> LDS copy. Async (ASYNCcnt-tracked, no VGPR round trip) when
// the gfx1250 builtin is available, else synchronous fallback.
__device__ __forceinline__ void copy16_to_lds(unsigned short* l, const unsigned short* g) {
#if HAVE_ASYNC
    __builtin_amdgcn_global_load_async_to_lds_b128(
        (v4i_vec*)(void*)g, (v4i_vec*)(void*)l, 0, 0);
#else
    *(uint4*)l = *(const uint4*)g;
#endif
}

__device__ __forceinline__ void wait_async() {
#if HAVE_ASYNC
#if __has_builtin(__builtin_amdgcn_s_wait_asynccnt)
    __builtin_amdgcn_s_wait_asynccnt(0);
#else
    asm volatile("s_wait_asynccnt 0x0" ::: "memory");
#endif
#endif
}

// ---------------------------------------------------------------------------
// fp32 -> bf16 casts
// ---------------------------------------------------------------------------
__global__ void cast_bf16_kernel(const float* __restrict__ in,
                                 unsigned short* __restrict__ out, size_t n) {
    size_t i = (size_t)blockIdx.x * blockDim.x + threadIdx.x;
    size_t stride = (size_t)gridDim.x * blockDim.x;
    for (; i < n; i += stride) out[i] = f2bf(in[i]);
}

// in: R x C row-major fp32 ; out: C x R row-major bf16 (= in transposed)
__global__ void transpose_cast_kernel(const float* __restrict__ in,
                                      unsigned short* __restrict__ out,
                                      int R, int C) {
    size_t total = (size_t)R * C;
    size_t i = (size_t)blockIdx.x * blockDim.x + threadIdx.x;
    size_t stride = (size_t)gridDim.x * blockDim.x;
    for (; i < total; i += stride) {
        size_t c = i / (size_t)R, r = i % (size_t)R;
        out[i] = f2bf(in[r * (size_t)C + c]);
    }
}

// ---------------------------------------------------------------------------
// bf16 WMMA GEMM: C[M,Nc] = A[M,K] * B[K,Nc] (+ bias[col])
// A: MxK row-major bf16. Bt: NcxK row-major bf16 (B transposed, K-contiguous).
// Block tile 128x128x64, 8 waves (2x4), wave tile 64x32.
// Double-buffered LDS, async global->LDS staging overlapped with WMMA compute.
// ---------------------------------------------------------------------------
#define SK 72   // LDS row stride in bf16 elements (64 + 8 pad)

__global__ __launch_bounds__(256)
void gemm_bf16_kernel(const unsigned short* __restrict__ A,
                      const unsigned short* __restrict__ Bt,
                      float* __restrict__ C,
                      const float* __restrict__ bias,
                      int M, int Nc, int K) {
    __shared__ __align__(16) unsigned short As[2][128 * SK];
    __shared__ __align__(16) unsigned short Bs[2][128 * SK];

    const int tid  = threadIdx.x;
    const int lane = tid & 31, wid = tid >> 5;
    const int lm = lane & 15, hi = lane >> 4;
    const int wave_m = (wid >> 2) * 64;
    const int wave_n = (wid & 3) * 32;
    const int m0 = blockIdx.y * 128;
    const int n0 = blockIdx.x * 128;

    v8f acc[4][2];
#pragma unroll
    for (int i = 0; i < 4; ++i)
#pragma unroll
        for (int j = 0; j < 2; ++j) acc[i][j] = v8f_zero();

    auto stage = [&](int buf, int kt) {
#pragma unroll
        for (int i = 0; i < 4; ++i) {
            int idx = tid + i * 256;                 // 0..1023
            int row = idx >> 3;
            int c8  = (idx & 7) * 8;
            copy16_to_lds(&As[buf][row * SK + c8],
                          &A[(size_t)(m0 + row) * K + kt + c8]);
            copy16_to_lds(&Bs[buf][row * SK + c8],
                          &Bt[(size_t)(n0 + row) * K + kt + c8]);
        }
    };

    const int nk = K / 64;
    stage(0, 0);
    for (int t = 0; t < nk; ++t) {
        const int cur = t & 1;
        wait_async();             // this wave's copies into buf cur are done
        __syncthreads();          // all waves' copies done; prev compute retired
        if (t + 1 < nk) stage(cur ^ 1, (t + 1) * 64);   // overlaps compute below

#pragma unroll
        for (int kc = 0; kc < 64; kc += 32) {
            Frag16 fa[4], fb[2];
#pragma unroll
            for (int mi = 0; mi < 4; ++mi) {
                // A frag: lanes 0-15 K runs {0-7,16-23}, lanes 16-31 {8-15,24-31}
                const unsigned short* ap =
                    &As[cur][(wave_m + mi * 16 + lm) * SK + kc + hi * 8];
                *(uint4*)&fa[mi].us[0] = *(const uint4*)ap;
                *(uint4*)&fa[mi].us[8] = *(const uint4*)(ap + 16);
            }
#pragma unroll
            for (int ni = 0; ni < 2; ++ni) {
                // B frag: column n = lm, lanes 0-15 K runs {0-7,16-23}, lanes 16-31 {8-15,24-31}
                const unsigned short* bp =
                    &Bs[cur][(wave_n + ni * 16 + lm) * SK + kc + hi * 8];
                *(uint4*)&fb[ni].us[0] = *(const uint4*)bp;
                *(uint4*)&fb[ni].us[8] = *(const uint4*)(bp + 16);
            }
#pragma unroll
            for (int mi = 0; mi < 4; ++mi)
#pragma unroll
                for (int ni = 0; ni < 2; ++ni)
                    acc[mi][ni] = __builtin_amdgcn_wmma_f32_16x16x32_bf16(
                        false, __builtin_bit_cast(v16bf, fa[mi]),
                        false, __builtin_bit_cast(v16bf, fb[ni]),
                        (short)0, acc[mi][ni], false, false);
        }
    }

    // epilogue: C/D layout -> row = base + r + 8*hi, col = lm
#pragma unroll
    for (int ni = 0; ni < 2; ++ni) {
        const int col = n0 + wave_n + ni * 16 + lm;
        const float bv = bias ? bias[col] : 0.f;
#pragma unroll
        for (int mi = 0; mi < 4; ++mi) {
            const int row0 = m0 + wave_m + mi * 16 + hi * 8;
#pragma unroll
            for (int r = 0; r < 8; ++r)
                C[(size_t)(row0 + r) * Nc + col] = acc[mi][ni][r] + bv;
        }
    }
}

// ---------------------------------------------------------------------------
// Pooling stage, split for grid-wide parallelism.
// ---------------------------------------------------------------------------
__global__ __launch_bounds__(256)
void pool_logits_kernel(const float* __restrict__ qkv,
                        const unsigned char* __restrict__ mask,
                        const float* __restrict__ w, int w_per_bh, int moff,
                        float* __restrict__ L) {
    __shared__ float wv[DHd];
    const int bh = blockIdx.x, chunk = blockIdx.y;
    const int b = bh >> 4, h = bh & 15;
    if (threadIdx.x < DHd)
        wv[threadIdx.x] = w[(w_per_bh ? bh * DHd : 0) + threadIdx.x];
    __syncthreads();

    const size_t rs = 3 * (size_t)INNERd;
    const float* base = qkv + (size_t)b * Ndim * rs + h * DHd + moff;
    const unsigned char* mrow = mask + (size_t)b * Ndim;
    const int n0 = chunk * (Ndim / 16);           // 512-row chunks
    for (int n = n0 + threadIdx.x; n < n0 + Ndim / 16; n += 256) {
        const float4* row = (const float4*)(base + (size_t)n * rs);
        float dot = 0.f;
#pragma unroll
        for (int j = 0; j < 16; ++j) {
            float4 v = row[j];
            dot += v.x * wv[4 * j] + v.y * wv[4 * j + 1] +
                   v.z * wv[4 * j + 2] + v.w * wv[4 * j + 3];
        }
        L[(size_t)bh * Ndim + n] = mrow[n] ? dot * 0.125f : MASK_VAL;
    }
}

__global__ __launch_bounds__(256)
void pool_softmax_kernel(float* __restrict__ L) {
    __shared__ float red[256];
    const int tid = threadIdx.x;
    float* Lr = L + (size_t)blockIdx.x * Ndim;

    float v[Ndim / 256];
    float lmax = -3.0e38f;
#pragma unroll
    for (int i = 0; i < Ndim / 256; ++i) {
        v[i] = Lr[tid + i * 256];
        lmax = fmaxf(lmax, v[i]);
    }
    red[tid] = lmax; __syncthreads();
    for (int s = 128; s > 0; s >>= 1) {
        if (tid < s) red[tid] = fmaxf(red[tid], red[tid + s]);
        __syncthreads();
    }
    const float mx = red[0]; __syncthreads();

    float sum = 0.f;
#pragma unroll
    for (int i = 0; i < Ndim / 256; ++i) {
        v[i] = __expf(v[i] - mx);
        sum += v[i];
    }
    red[tid] = sum; __syncthreads();
    for (int s = 128; s > 0; s >>= 1) {
        if (tid < s) red[tid] += red[tid + s];
        __syncthreads();
    }
    const float inv = 1.f / red[0];
#pragma unroll
    for (int i = 0; i < Ndim / 256; ++i) Lr[tid + i * 256] = v[i] * inv;
}

__global__ __launch_bounds__(256)
void pool_wsum_kernel(const float* __restrict__ qkv,
                      const float* __restrict__ L, int moff,
                      float* __restrict__ part) {
    __shared__ float red[256];
    const int bh = blockIdx.x, chunk = blockIdx.y;
    const int b = bh >> 4, h = bh & 15;
    const size_t rs = 3 * (size_t)INNERd;
    const float* base = qkv + (size_t)b * Ndim * rs + h * DHd + moff;
    const float* Lr = L + (size_t)bh * Ndim;
    const int d = threadIdx.x & 63, g = threadIdx.x >> 6;
    const int n0 = chunk * (Ndim / 16);
    float acc = 0.f;
    for (int n = n0 + g; n < n0 + Ndim / 16; n += 4)
        acc += Lr[n] * base[(size_t)n * rs + d];
    red[threadIdx.x] = acc; __syncthreads();
    if (threadIdx.x < DHd)
        part[((size_t)bh * 16 + chunk) * DHd + d] =
            red[d] + red[64 + d] + red[128 + d] + red[192 + d];
}

__global__ __launch_bounds__(64)
void pool_reduce_kernel(const float* __restrict__ part,
                        float* __restrict__ gout,
                        const float* __restrict__ wk,      // null on k-pass
                        float* __restrict__ wk_eff) {      // null on k-pass
    const int bh = blockIdx.x, d = threadIdx.x;
    float s = 0.f;
#pragma unroll
    for (int c = 0; c < 16; ++c) s += part[((size_t)bh * 16 + c) * DHd + d];
    gout[(size_t)bh * DHd + d] = s;
    if (wk) wk_eff[(size_t)bh * DHd + d] = wk[d] * s;
}

// ---------------------------------------------------------------------------
// Per-head: r = (v * global_k) @ W_r + b_r + q  -> bf16
// Block-diagonal WMMA GEMM, K=64, per-wave 16x64 output tile.
// ---------------------------------------------------------------------------
__global__ __launch_bounds__(256)
void head_kernel(const float* __restrict__ qkv,
                 const float* __restrict__ gk,
                 const unsigned short* __restrict__ wr_t,  // [64][64] = W_r^T bf16
                 const float* __restrict__ b_r,
                 unsigned short* __restrict__ r_out) {
    __shared__ float gks[DHd];
    __shared__ float brs[DHd];

    const int tid  = threadIdx.x;
    const int lane = tid & 31, wid = tid >> 5;
    const int lm = lane & 15, hi = lane >> 4;
    const int h = blockIdx.y;
    const int m0 = blockIdx.x * 128 + wid * 16;      // global row base (b*N+n)
    const int b = (blockIdx.x * 128) / Ndim;

    if (tid < DHd) {
        gks[tid] = gk[((size_t)b * Hd + h) * DHd + tid];
        brs[tid] = b_r[tid];
    }
    __syncthreads();

    const size_t rs = 3 * (size_t)INNERd;
    const float* qbase = qkv + h * DHd;
    const float* vbase = qkv + 2 * INNERd + h * DHd;

    v8f acc[4];
#pragma unroll
    for (int ni = 0; ni < 4; ++ni) {
        const int col = ni * 16 + lm;
        const float bb = brs[col];
        v8f a;
#pragma unroll
        for (int r = 0; r < 8; ++r) {
            int row = m0 + r + hi * 8;
            a[r] = qbase[(size_t)row * rs + col] + bb;
        }
        acc[ni] = a;
    }

#pragma unroll
    for (int kc = 0; kc < 64; kc += 32) {
        Frag16 fa;
        const int row = m0 + lm;
        const int s1 = kc + hi * 8;
        const float* vr = vbase + (size_t)row * rs;
#pragma unroll
        for (int run = 0; run < 2; ++run) {
            const int s = s1 + run * 16;
            const float4 f0 = *(const float4*)(vr + s);
            const float4 f1 = *(const float4*)(vr + s + 4);
            fa.us[run * 8 + 0] = f2bf(f0.x * gks[s + 0]);
            fa.us[run * 8 + 1] = f2bf(f0.y * gks[s + 1]);
            fa.us[run * 8 + 2] = f2bf(f0.z * gks[s + 2]);
            fa.us[run * 8 + 3] = f2bf(f0.w * gks[s + 3]);
            fa.us[run * 8 + 4] = f2bf(f1.x * gks[s + 4]);
            fa.us[run * 8 + 5] = f2bf(f1.y * gks[s + 5]);
            fa.us[run * 8 + 6] = f2bf(f1.z * gks[s + 6]);
            fa.us[run * 8 + 7] = f2bf(f1.w * gks[s + 7]);
        }
#pragma unroll
        for (int ni = 0; ni < 4; ++ni) {
            Frag16 fb;
            const unsigned short* bp = wr_t + (ni * 16 + lm) * DHd + kc + hi * 8;
            *(uint4*)&fb.us[0] = *(const uint4*)bp;
            *(uint4*)&fb.us[8] = *(const uint4*)(bp + 16);
            acc[ni] = __builtin_amdgcn_wmma_f32_16x16x32_bf16(
                false, __builtin_bit_cast(v16bf, fa),
                false, __builtin_bit_cast(v16bf, fb),
                (short)0, acc[ni], false, false);
        }
    }

#pragma unroll
    for (int ni = 0; ni < 4; ++ni) {
        const int col = ni * 16 + lm;
#pragma unroll
        for (int r = 0; r < 8; ++r) {
            int row = m0 + r + hi * 8;
            r_out[(size_t)row * INNERd + h * DHd + col] = f2bf(acc[ni][r]);
        }
    }
}

// ---------------------------------------------------------------------------
extern "C" void kernel_launch(void* const* d_in, const int* in_sizes, int n_in,
                              void* d_out, int out_size, void* d_ws, size_t ws_size,
                              hipStream_t stream) {
    const float*         x     = (const float*)d_in[0];
    const unsigned char* mask  = (const unsigned char*)d_in[1];
    const float*         W_qkv = (const float*)d_in[2];
    const float*         w_q   = (const float*)d_in[3];
    const float*         w_k   = (const float*)d_in[4];
    const float*         W_r   = (const float*)d_in[5];
    const float*         b_r   = (const float*)d_in[6];
    const float*         W_out = (const float*)d_in[7];
    const float*         b_out = (const float*)d_in[8];
    float* out = (float*)d_out;

    const size_t M = (size_t)Bdim * Ndim;      // 32768
    char* ws = (char*)d_ws;
    size_t off = 0;
    auto take = [&](size_t bytes) -> char* {
        char* p = ws + off;
        off = (off + bytes + 255) & ~(size_t)255;
        return p;
    };
    unsigned short* xbf   = (unsigned short*)take(M * DIMd * 2);
    unsigned short* wqkvt = (unsigned short*)take((size_t)3 * INNERd * DIMd * 2);
    unsigned short* wrt   = (unsigned short*)take((size_t)DHd * DHd * 2);
    unsigned short* woutt = (unsigned short*)take((size_t)DIMd * INNERd * 2);
    float*          qkv   = (float*)take(M * 3 * INNERd * 4);
    float*          Lbuf  = (float*)take((size_t)Bdim * Hd * Ndim * 4);
    float*          part  = (float*)take((size_t)Bdim * Hd * 16 * DHd * 4);
    float*          gqbuf = (float*)take((size_t)Bdim * Hd * DHd * 4);
    float*          wkeff = (float*)take((size_t)Bdim * Hd * DHd * 4);
    float*          gkbuf = (float*)take((size_t)Bdim * Hd * DHd * 4);
    unsigned short* rbf   = (unsigned short*)take(M * INNERd * 2);

    // 1) casts / transposed weight casts
    cast_bf16_kernel<<<4096, 256, 0, stream>>>(x, xbf, M * DIMd);
    transpose_cast_kernel<<<2048, 256, 0, stream>>>(W_qkv, wqkvt, DIMd, 3 * INNERd);
    transpose_cast_kernel<<<16, 256, 0, stream>>>(W_r, wrt, DHd, DHd);
    transpose_cast_kernel<<<1024, 256, 0, stream>>>(W_out, woutt, INNERd, DIMd);

    // 2) qkv = x @ W_qkv
    gemm_bf16_kernel<<<dim3(3 * INNERd / 128, (int)(M / 128)), 256, 0, stream>>>(
        xbf, wqkvt, qkv, nullptr, (int)M, 3 * INNERd, DIMd);

    // 3) q-pool -> global_q -> gated weights
    pool_logits_kernel<<<dim3(Bdim * Hd, 16), 256, 0, stream>>>(
        qkv, mask, w_q, 0, 0, Lbuf);
    pool_softmax_kernel<<<Bdim * Hd, 256, 0, stream>>>(Lbuf);
    pool_wsum_kernel<<<dim3(Bdim * Hd, 16), 256, 0, stream>>>(qkv, Lbuf, 0, part);
    pool_reduce_kernel<<<Bdim * Hd, 64, 0, stream>>>(part, gqbuf, w_k, wkeff);

    // 4) gated k-pool -> global_k
    pool_logits_kernel<<<dim3(Bdim * Hd, 16), 256, 0, stream>>>(
        qkv, mask, wkeff, 1, INNERd, Lbuf);
    pool_softmax_kernel<<<Bdim * Hd, 256, 0, stream>>>(Lbuf);
    pool_wsum_kernel<<<dim3(Bdim * Hd, 16), 256, 0, stream>>>(qkv, Lbuf, INNERd, part);
    pool_reduce_kernel<<<Bdim * Hd, 64, 0, stream>>>(part, gkbuf, nullptr, nullptr);

    // 5) r = (v*gk) @ W_r + b_r + q  -> bf16
    head_kernel<<<dim3((int)(M / 128), Hd), 256, 0, stream>>>(qkv, gkbuf, wrt, b_r, rbf);

    // 6) out = r @ W_out + b_out
    gemm_bf16_kernel<<<dim3(DIMd / 128, (int)(M / 128)), 256, 0, stream>>>(
        rbf, woutt, out, b_out, (int)M, DIMd, INNERd);
}
